// DSVF_63144609185910
// MI455X (gfx1250) — matrix operand order repaired and checked by reading, not compiled, not run on hardware
//
#include <hip/hip_runtime.h>
#include <hip/hip_bf16.h>
#include <math.h>

// ---------------------------------------------------------------------------
// DSVF via exact circular-convolution GEMM, bf16-split (hi/lo) on the
// V_WMMA_F32_16X16X32_BF16 pipe (gfx1250).
//
// out[b, s*2048+n] = sum_{q=0}^{4095} x[b,(s-1)*2048+q] * h[(n+2048-q) & 4095]
// h = irfft( rfft(b,4096)/rfft(a,4096) ), computed in closed form.
// A = x rows (16 batches), B = circulant of h, K = 4096, split into
// Ahi*Bhi + Ahi*Blo + Alo*Bhi  (bf16x3, ~3e-5 relative error).
// Tables staged into LDS with GLOBAL_LOAD_ASYNC_TO_LDS_B128 (ASYNCcnt).
// ---------------------------------------------------------------------------

typedef __attribute__((ext_vector_type(8)))  float  v8f;
typedef __attribute__((ext_vector_type(16))) __bf16 v16bf;
typedef __attribute__((ext_vector_type(8)))  __bf16 v8bf;

#define LROW   262144   // samples per batch row
#define SEGN   2048     // segment length N
#define NFFT   4096
#define NSEG   128      // segments per row
#define NBATCH 64

#define CPY    4128               // table copy length (4096 + 32 wrap margin)
#define TBL_HALF (8 * CPY)        // 33024 elements per precision (8 copies)
#define TBL_ELEMS (2 * TBL_HALF)  // 66048 bf16 = 132096 bytes
#define TBL_BYTES (TBL_ELEMS * 2)
#define SMEM_BYTES TBL_BYTES

// ------------------------- Kernel 1: H(k) = B(k)/A(k) ----------------------
__global__ void DSVF_build_H(const float* __restrict__ gp,
                             const float* __restrict__ Rp,
                             const float* __restrict__ mhp,
                             const float* __restrict__ mbp,
                             const float* __restrict__ mlp,
                             float* __restrict__ Hr,
                             float* __restrict__ Hi) {
    int k = blockIdx.x * blockDim.x + threadIdx.x;
    if (k > NFFT / 2) return;

    const float PI_F = 3.14159265358979323846f;
    float sig = 1.0f / (1.0f + expf(-gp[0]));
    float g   = tanf(PI_F * sig * 0.5f);
    float R   = log1pf(expf(Rp[0]));          // softplus
    float g2  = g * g;
    float mh = mhp[0], mb = mbp[0], ml = mlp[0];

    float b0 = g2 * ml + g * mb + mh;
    float b1 = 2.0f * g2 * ml - 2.0f * mh;
    float b2 = g2 * ml - g * mb + mh;
    float a0 = g2 + 2.0f * R * g + 1.0f;
    float a1 = 2.0f * g2 - 2.0f;
    float a2 = g2 - 2.0f * R * g + 1.0f;

    double th = (double)k * (2.0 * M_PI / (double)NFFT);
    double c1 = cos(th),       s1 = -sin(th);
    double c2 = cos(2.0 * th), s2 = -sin(2.0 * th);

    double Br = (double)b0 + (double)b1 * c1 + (double)b2 * c2;
    double Bi =              (double)b1 * s1 + (double)b2 * s2;
    double Ar = (double)a0 + (double)a1 * c1 + (double)a2 * c2;
    double Ai =              (double)a1 * s1 + (double)a2 * s2;
    double den = Ar * Ar + Ai * Ai;

    Hr[k] = (float)((Br * Ar + Bi * Ai) / den);
    Hi[k] = (float)((Bi * Ar - Br * Ai) / den);
}

// ------------------- Kernel 2: h[n] = irfft(H) (direct iDFT) ---------------
__global__ void DSVF_build_h(const float* __restrict__ Hr,
                             const float* __restrict__ Hi,
                             float* __restrict__ h) {
    int n = blockIdx.x * blockDim.x + threadIdx.x;   // 0..4095
    if (n >= NFFT) return;

    double acc = (double)Hr[0] + (double)Hr[NFFT / 2] * ((n & 1) ? -1.0 : 1.0);
    const double w = 2.0 * M_PI / (double)NFFT;
    for (int k = 1; k < NFFT / 2; ++k) {
        int p = (k * n) & (NFFT - 1);                // exact integer phase reduction
        double th = (double)p * w;
        acc += 2.0 * ((double)Hr[k] * cos(th) - (double)Hi[k] * sin(th));
    }
    h[n] = (float)(acc * (1.0 / (double)NFFT));
}

// ---- Kernel 3: 8 alignment-shifted bf16 hi/lo copies of reversed h --------
// T_hi[a][t] = bf16_hi( h[(2048 - a - t) & 4095] ), a = 0..7, t = 0..4127
// T_lo likewise with the bf16 residual. Layout: [hi tables][lo tables].
__global__ void DSVF_build_tables(const float* __restrict__ h,
                                  __bf16* __restrict__ gT) {
    int i = blockIdx.x * blockDim.x + threadIdx.x;
    if (i >= TBL_ELEMS) return;
    int j  = (i >= TBL_HALF) ? (i - TBL_HALF) : i;
    int a  = j / CPY;
    int t  = j - a * CPY;
    int idx = (SEGN - a - t) & (NFFT - 1);
    float  hv = h[idx];
    __bf16 hi = (__bf16)hv;
    gT[i] = (i >= TBL_HALF) ? (__bf16)(hv - (float)hi) : hi;
}

// -------- Kernel 4: circulant GEMM, bf16x3 on V_WMMA_F32_16X16X32_BF16 -----
// grid: (128 segments, 4 batch-groups), 256 threads = 8 waves.
// K outer (chunks of 32); each wave keeps 16 accumulator tiles (16x16) live
// so the A fragment (global loads + hi/lo split) is built once per chunk.
__launch_bounds__(256)
__global__ void DSVF_conv_wmma(const float* __restrict__ x,
                               const __bf16* __restrict__ gT,
                               float* __restrict__ out) {
    extern __shared__ __align__(16) __bf16 sT[];     // 132 KB dynamic LDS

    const int s   = blockIdx.x;                      // segment 0..127
    const int bg  = blockIdx.y;                      // batch group 0..3
    const int tid = threadIdx.x;

    // Stage hrev hi/lo alignment tables: async memory -> LDS copy, no VGPR
    // round-trip; 16B per lane per op, drained once on ASYNCcnt.
    {
        unsigned    sbase = (unsigned)(size_t)(void*)sT;  // LDS byte offset
        const char* gbase = (const char*)gT;
        for (int i = tid * 16; i < TBL_BYTES; i += 256 * 16) {
            unsigned    laddr = sbase + i;
            const void* ga    = gbase + i;
            asm volatile("global_load_async_to_lds_b128 %0, %1, off"
                         :: "v"(laddr), "v"(ga) : "memory");
        }
        asm volatile("s_wait_asynccnt 0x0" ::: "memory");
    }
    __syncthreads();

    const int lane = tid & 31;
    const int wave = tid >> 5;                       // 0..7
    const int m    = lane & 15;                      // A row (batch in group)
    const int half = lane >> 4;                      // K sub-block select
    const int n    = m;                              // B column within tile

    const int b0      = bg * 16;
    const int k_start = (s == 0) ? SEGN : 0;         // segment 0: zero prefix

    // Per-lane fixed alignment copy: o mod 8 == (-n) mod 8 always.
    const int acopy = (16 - n) & 7;
    const char* pThi = (const char*)(sT + acopy * CPY);
    const char* pTlo = (const char*)(sT + TBL_HALF + acopy * CPY);
    const int  half32 = half * 16;                   // byte offset, lanes 16-31

    // A row pointer at k = k_start (no negative intermediate for s==0).
    const float* __restrict__ ap =
        x + (long)(b0 + m) * LROW + (long)(s - 1) * SEGN + half * 8 + k_start;

    v8f acc[16];
    #pragma unroll
    for (int t = 0; t < 16; ++t)
        acc[t] = (v8f){0.f, 0.f, 0.f, 0.f, 0.f, 0.f, 0.f, 0.f};

    for (int k0 = k_start; k0 < NFFT; k0 += 32, ap += 32) {
        // ---- A fragment: 16 f32 -> bf16 hi/lo (layout: K = half*8+{0..7},
        //      16+half*8+{0..7} packed pairwise into 8 dwords) ----
        float4 f0 = *(const float4*)(ap);
        float4 f1 = *(const float4*)(ap + 4);
        float4 f2 = *(const float4*)(ap + 16);
        float4 f3 = *(const float4*)(ap + 20);

        v16bf Ahi, Alo;
        #define CVT4(F, B)                                                  \
            { __bf16 h0 = (__bf16)F.x, h1 = (__bf16)F.y,                    \
                     h2 = (__bf16)F.z, h3 = (__bf16)F.w;                    \
              Ahi[B+0] = h0; Ahi[B+1] = h1; Ahi[B+2] = h2; Ahi[B+3] = h3;   \
              Alo[B+0] = (__bf16)(F.x - (float)h0);                         \
              Alo[B+1] = (__bf16)(F.y - (float)h1);                         \
              Alo[B+2] = (__bf16)(F.z - (float)h2);                         \
              Alo[B+3] = (__bf16)(F.w - (float)h3); }
        CVT4(f0, 0) CVT4(f1, 4) CVT4(f2, 8) CVT4(f3, 12)
        #undef CVT4

        const int kmn = k0 - n;                      // o = (k0 - n0 - n) mod 4096
        #pragma unroll
        for (int t = 0; t < 16; ++t) {
            const int n0 = (wave + t * 8) << 4;
            const int o  = (kmn - n0) & (NFFT - 1);
            const int bo = ((o >> 3) << 4) + half32; // (o - acopy)*2 bytes, 16B-aligned

            v8bf bh0 = *(const v8bf*)(pThi + bo);
            v8bf bh1 = *(const v8bf*)(pThi + bo + 32);
            v8bf bl0 = *(const v8bf*)(pTlo + bo);
            v8bf bl1 = *(const v8bf*)(pTlo + bo + 32);
            v16bf Bh = __builtin_shufflevector(bh0, bh1,
                        0,1,2,3,4,5,6,7,8,9,10,11,12,13,14,15);
            v16bf Bl = __builtin_shufflevector(bl0, bl1,
                        0,1,2,3,4,5,6,7,8,9,10,11,12,13,14,15);

            acc[t] = __builtin_amdgcn_wmma_f32_16x16x32_bf16(
                         false, Alo, false, Bh, (short)0, acc[t], false, false);
            acc[t] = __builtin_amdgcn_wmma_f32_16x16x32_bf16(
                         false, Ahi, false, Bl, (short)0, acc[t], false, false);
            acc[t] = __builtin_amdgcn_wmma_f32_16x16x32_bf16(
                         false, Ahi, false, Bh, (short)0, acc[t], false, false);
        }
    }

    // C/D layout: VGPR r -> row M=r (lanes 0-15) / M=r+8 (lanes 16-31), N=lane&15
    // Output is write-once / never re-read: non-temporal stores keep L2 for
    // the x-rows and tables that do have reuse.
    #pragma unroll
    for (int t = 0; t < 16; ++t) {
        const int n0 = (wave + t * 8) << 4;
        float* op = out + (long)(b0 + half * 8) * LROW + (long)s * SEGN + n0 + n;
        #pragma unroll
        for (int r = 0; r < 8; ++r)
            __builtin_nontemporal_store(acc[t][r], op + (long)r * LROW);
    }
}

// ---------------------------------------------------------------------------
extern "C" void kernel_launch(void* const* d_in, const int* in_sizes, int n_in,
                              void* d_out, int out_size, void* d_ws, size_t ws_size,
                              hipStream_t stream) {
    const float* x   = (const float*)d_in[0];
    const float* gp  = (const float*)d_in[1];
    const float* Rp  = (const float*)d_in[2];
    const float* mhp = (const float*)d_in[3];
    const float* mbp = (const float*)d_in[4];
    const float* mlp = (const float*)d_in[5];
    float* out = (float*)d_out;

    // ws layout (floats): Hr @0, Hi @4096, h @8192, bf16 tables @12288
    float*  Hr = (float*)d_ws;
    float*  Hi = Hr + 4096;
    float*  h  = Hr + 8192;
    __bf16* gT = (__bf16*)(Hr + 12288);

    DSVF_build_H<<<dim3((NFFT / 2 + 256) / 256), dim3(256), 0, stream>>>(
        gp, Rp, mhp, mbp, mlp, Hr, Hi);
    DSVF_build_h<<<dim3(NFFT / 256), dim3(256), 0, stream>>>(Hr, Hi, h);
    DSVF_build_tables<<<dim3((TBL_ELEMS + 255) / 256), dim3(256), 0, stream>>>(h, gT);

    dim3 grid(NSEG, NBATCH / 16);                    // 128 x 4 workgroups
    DSVF_conv_wmma<<<grid, dim3(256), SMEM_BYTES, stream>>>(x, gT, out);
}
